// GCN_68659347193894
// MI455X (gfx1250) — matrix operand order repaired and checked by reading, not compiled, not run on hardware
//
#include <hip/hip_runtime.h>
#include <hip/hip_bf16.h>

typedef __attribute__((ext_vector_type(16))) _Float16 v16h;
typedef __attribute__((ext_vector_type(8)))  float    v8f;

static constexpr int NN = 20000;    // nodes
static constexpr int NE = 320000;   // edges (without self loops)
static constexpr int ET = NE + NN;  // edges + self loops

__device__ __forceinline__ void atomAddF32(float* p, float v) {
    // lowers to global_atomic_add_f32 (relaxed, device scope) on gfx1250
    __hip_atomic_fetch_add(p, v, __ATOMIC_RELAXED, __HIP_MEMORY_SCOPE_AGENT);
}

// ---------------- graph normalization ----------------
__global__ void kzero(float* p, int n) {
    int i = blockIdx.x * blockDim.x + threadIdx.x;
    if (i < n) p[i] = 0.f;
}

__global__ void kdeg(const int* __restrict__ ei, const float* __restrict__ ew,
                     float* __restrict__ deg) {
    int e = blockIdx.x * blockDim.x + threadIdx.x;
    if (e >= ET) return;
    int d; float w;
    if (e < NE) { d = ei[NE + e]; w = ew[e]; }
    else        { d = e - NE;     w = 1.f;   }
    atomAddF32(&deg[d], w);
}

__global__ void knorm(const int* __restrict__ ei, const float* __restrict__ ew,
                      const float* __restrict__ deg, float* __restrict__ norm) {
    int e = blockIdx.x * blockDim.x + threadIdx.x;
    if (e >= ET) return;
    int s, d; float w;
    if (e < NE) { s = ei[e]; d = ei[NE + e]; w = ew[e]; }
    else        { s = d = e - NE;            w = 1.f;   }
    float ds = deg[s], dd = deg[d];
    float a = ds > 0.f ? rsqrtf(ds) : 0.f;
    float b = dd > 0.f ? rsqrtf(dd) : 0.f;
    norm[e] = a * w * b;
}

// -------- weight convert+transpose: Wt[c][k] = (f16) W[k][c] --------
__global__ void kcvtW(const float* __restrict__ W, _Float16* __restrict__ Wt,
                      int Cin, int Cout) {
    int i = blockIdx.x * blockDim.x + threadIdx.x;
    if (i >= Cin * Cout) return;
    int k = i / Cout, c = i % Cout;
    Wt[(size_t)c * Cin + k] = (_Float16)W[i];
}

// ---------------- WMMA GEMM: C[NN x Cout] = A[NN x Cin] @ W ----------------
// One wave32 computes a 16 x (NB*16) output strip: the A fragment is loaded
// once per K-step and reused across NB=4 column tiles (4 independent
// accumulator chains -> 4x less A traffic, WMMA RAW latency hidden).
// K loop of 32 through v_wmma_f32_16x16x32_f16 (f32 accumulate). A is
// converted f32->f16 in regs in contiguous k order (k = 16*(lane/16) + s), same as B; B is
// pre-transposed/converted so a lane's fragment is one contiguous 32B load.
__global__ void kgemm_wmma(const float* __restrict__ A,
                           const _Float16* __restrict__ Bt,
                           float* __restrict__ C, int Cin, int Cout) {
    constexpr int NB = 4;                    // column tiles per wave
    int wave = (int)((blockIdx.x * blockDim.x + threadIdx.x) >> 5);
    int lane = threadIdx.x & 31;
    int groupsN = (Cout >> 4) / NB;          // Cout/16 is 4,8,16 -> divisible
    int strips  = (NN >> 4) * groupsN;
    if (wave >= strips) return;              // wave-uniform: EXEC all-1s inside
    int tm = wave / groupsN, tg = wave % groupsN;
    int row0 = tm << 4, col0 = tg * (NB << 4);
    int m  = lane & 15;                      // A row / B col / C col within tile
    int hi = (lane >> 4) << 3;               // 0 (lanes 0-15) or 8 (lanes 16-31)

    const float*    Arow = A  + (size_t)(row0 + m) * Cin;
    const _Float16* Bcol = Bt + (size_t)(col0 + m) * Cin + (hi << 1); // kb = 0/16
    const size_t    Bstride = (size_t)16 * Cin;                       // next tile

    v8f acc[NB] = {};
    for (int k0 = 0; k0 < Cin; k0 += 32) {
        v16h a;
        const float* ap = Arow + k0 + (hi << 1); // K = k0+2*hi+0..15
        #pragma unroll
        for (int j = 0; j < 8; ++j) {
            a[j]     = (_Float16)ap[j];
            a[8 + j] = (_Float16)ap[8 + j];
        }
        #pragma unroll
        for (int t = 0; t < NB; ++t) {
            v16h b = *(const v16h*)(Bcol + (size_t)t * Bstride + k0); // 32B load
            acc[t] = __builtin_amdgcn_wmma_f32_16x16x32_f16(
                         false, a, false, b, (short)0, acc[t], false, false);
        }
    }
    // C/D layout: VGPR v holds row M = v + hi, col = lane&15
    #pragma unroll
    for (int t = 0; t < NB; ++t) {
        float* Crow = C + (size_t)row0 * Cout + col0 + (t << 4) + m;
        #pragma unroll
        for (int v = 0; v < 8; ++v)
            Crow[(size_t)(v + hi) * Cout] = acc[t][v];
    }
}

// ---------------- layer-5 matvec (Cout = 1) ----------------
__global__ void kmatvec(const float* __restrict__ A, const float* __restrict__ W,
                        float* __restrict__ y, int Cin) {
    int n = blockIdx.x * blockDim.x + threadIdx.x;
    if (n >= NN) return;
    const float* a = A + (size_t)n * Cin;
    float s = 0.f;
    for (int k = 0; k < Cin; ++k) s += a[k] * W[k];
    y[n] = s;
}

// ---------------- bias init / scatter / activations ----------------
__global__ void kbias(float* __restrict__ out, const float* __restrict__ b,
                      int total, int Cout) {
    int i = blockIdx.x * blockDim.x + threadIdx.x;
    if (i < total) out[i] = b[i % Cout];
}

__global__ void kagg4(const int* __restrict__ ei, const float* __restrict__ norm,
                      const float* __restrict__ Hd, float* __restrict__ out,
                      int Cout) {
    int chunks = Cout >> 2;
    long long gid = (long long)blockIdx.x * blockDim.x + threadIdx.x;
    long long tot = (long long)ET * chunks;
    if (gid >= tot) return;
    int e = (int)(gid / chunks);
    int q = (int)(gid % chunks);
    int s, d;
    if (e < NE) { s = ei[e]; d = ei[NE + e]; }
    else        { s = d = e - NE; }
    float w = norm[e];
    const float4 v = *(const float4*)(Hd + (size_t)s * Cout + (q << 2));
    float* o = out + (size_t)d * Cout + (q << 2);
    atomAddF32(o + 0, w * v.x);
    atomAddF32(o + 1, w * v.y);
    atomAddF32(o + 2, w * v.z);
    atomAddF32(o + 3, w * v.w);
}

__global__ void kagg1(const int* __restrict__ ei, const float* __restrict__ norm,
                      const float* __restrict__ Hd, float* __restrict__ out) {
    int e = blockIdx.x * blockDim.x + threadIdx.x;
    if (e >= ET) return;
    int s, d;
    if (e < NE) { s = ei[e]; d = ei[NE + e]; }
    else        { s = d = e - NE; }
    atomAddF32(&out[d], norm[e] * Hd[s]);
}

__global__ void krelu(float* p, int n) {
    int i = blockIdx.x * blockDim.x + threadIdx.x;
    if (i < n) p[i] = fmaxf(p[i], 0.f);
}

__global__ void ktanhout(const float* __restrict__ in, float* __restrict__ out, int n) {
    int i = blockIdx.x * blockDim.x + threadIdx.x;
    if (i < n) out[i] = tanhf(in[i]);
}

// ---------------- host driver ----------------
extern "C" void kernel_launch(void* const* d_in, const int* in_sizes, int n_in,
                              void* d_out, int out_size, void* d_ws, size_t ws_size,
                              hipStream_t stream) {
    (void)in_sizes; (void)n_in; (void)out_size; (void)ws_size;
    const float* x  = (const float*)d_in[0];
    const int*   ei = (const int*)  d_in[1];
    const float* ew = (const float*)d_in[2];
    const float* Wl[5] = { (const float*)d_in[3], (const float*)d_in[5],
                           (const float*)d_in[7], (const float*)d_in[9],
                           (const float*)d_in[11] };
    const float* bl[5] = { (const float*)d_in[4], (const float*)d_in[6],
                           (const float*)d_in[8], (const float*)d_in[10],
                           (const float*)d_in[12] };
    const int dims[6] = {128, 64, 128, 256, 64, 1};

    // workspace carve-up (256B aligned)
    size_t off = 0;
    auto carve = [&](size_t bytes) -> char* {
        char* p = (char*)d_ws + off;
        off = (off + bytes + 255) & ~(size_t)255;
        return p;
    };
    float*    norm = (float*)   carve((size_t)ET * 4);
    float*    deg  = (float*)   carve((size_t)NN * 4);
    float*    Hd   = (float*)   carve((size_t)NN * 256 * 4);
    float*    HA   = (float*)   carve((size_t)NN * 256 * 4);
    float*    HB   = (float*)   carve((size_t)NN * 256 * 4);
    _Float16* Wt   = (_Float16*)carve((size_t)256 * 256 * 2);

    const int B = 256;
    auto nb = [](long long n, int b) { return (int)((n + b - 1) / b); };

    // gcn_norm
    kzero<<<nb(NN, B), B, 0, stream>>>(deg, NN);
    kdeg <<<nb(ET, B), B, 0, stream>>>(ei, ew, deg);
    knorm<<<nb(ET, B), B, 0, stream>>>(ei, ew, deg, norm);

    const float* Hin = x;
    float* bufs[2] = { HA, HB };
    for (int L = 0; L < 4; ++L) {
        int Cin = dims[L], Cout = dims[L + 1];
        float* Hout = bufs[L & 1];
        // dense GEMM through WMMA (NB=4 column tiles per wave)
        kcvtW<<<nb((long long)Cin * Cout, B), B, 0, stream>>>(Wl[L], Wt, Cin, Cout);
        long long waves = (long long)(NN / 16) * ((Cout / 16) / 4);
        kgemm_wmma<<<nb(waves * 32, B), B, 0, stream>>>(Hin, Wt, Hd, Cin, Cout);
        // out = bias; out[dst] += norm * Hd[src]; relu
        kbias<<<nb((long long)NN * Cout, B), B, 0, stream>>>(Hout, bl[L], NN * Cout, Cout);
        kagg4<<<nb((long long)ET * (Cout / 4), B), B, 0, stream>>>(ei, norm, Hd, Hout, Cout);
        krelu<<<nb((long long)NN * Cout, B), B, 0, stream>>>(Hout, NN * Cout);
        Hin = Hout;
    }

    // layer 5: Cin=64, Cout=1 -> matvec + scalar scatter + tanh
    float* out5 = HA;   // layer-3 buffer is free by now
    kmatvec<<<nb(NN, B), B, 0, stream>>>(Hin, Wl[4], Hd, 64);
    kbias  <<<nb(NN, B), B, 0, stream>>>(out5, bl[4], NN, 1);
    kagg1  <<<nb(ET, B), B, 0, stream>>>(ei, norm, Hd, out5);
    ktanhout<<<nb(NN, B), B, 0, stream>>>(out5, (float*)d_out, NN);
}
